// DialogueEmpathyModel_19232863551886
// MI455X (gfx1250) — matrix operand order repaired and checked by reading, not compiled, not run on hardware
//
#include <hip/hip_runtime.h>

// ---------------------------------------------------------------------------
// DialogueEmpathyModel for MI455X (gfx1250, wave32, WMMA).
// Output depends only on: x -> global GRU (selected speaker) -> emotion GRU ->
// head. Party GRU + attention are dead code and eliminated.
// Pipeline:
//   1) pack x (B,S,D)->(S,B,D) bf16; convert recurrent weights to bf16
//   2) bulk WMMA GEMM: gi_gx = Xbf @ gg_wih[:, :D].T + gg_bih      (S*B x 3H)
//   3) persistent kernel: global GRU recurrence (M=16 WMMA tiles, state in LDS)
//   4) bulk WMMA GEMM: gi_es = spk_state_bf @ eg_wih[:, H:].T + eg_bih
//   5) persistent kernel: emotion GRU + output head -> scores (B,S,1)
// ---------------------------------------------------------------------------

#define BB  16
#define SS  512
#define DD  1024
#define HH  512
#define SPN 8
#define H3  1536

typedef __attribute__((ext_vector_type(16))) __bf16 v16bf;
typedef __attribute__((ext_vector_type(8)))  float  v8f;
typedef unsigned uv4 __attribute__((ext_vector_type(4)));

union Frag { v16bf v; uv4 q[2]; };

__device__ __forceinline__ unsigned short f2bf(float x) {
  unsigned u = __float_as_uint(x);
  unsigned r = u + 0x7FFFu + ((u >> 16) & 1u);   // round-to-nearest-even
  return (unsigned short)(r >> 16);
}
__device__ __forceinline__ float sigm(float x) { return 1.f / (1.f + __expf(-x)); }

// One 16x16xK WMMA dot: C(16x16) += A(16xK) * W(16xK)^T, bf16 in / f32 out.
// arow: A row (lane%16) base ptr, K contiguous bf16. wrow: W row (lane%16) base.
__device__ __forceinline__ v8f wmma_dot(const unsigned short* arow,
                                        const unsigned short* wrow,
                                        int K, int lane) {
  const int khA = (lane >= 16) ? 8  : 0;   // A: K 0..7/16..23 vs 8..15/24..31
  const int kB  = (lane >= 16) ? 8  : 0;   // B: K 0..7/16..23 vs 8..15/24..31
  v8f acc = {};
  for (int k0 = 0; k0 < K; k0 += 32) {
    Frag a, b;
    a.q[0] = *(const uv4*)(arow + k0 + khA);
    a.q[1] = *(const uv4*)(arow + k0 + 16 + khA);
    b.q[0] = *(const uv4*)(wrow + k0 + kB);
    b.q[1] = *(const uv4*)(wrow + k0 + 16 + kB);
    acc = __builtin_amdgcn_wmma_f32_16x16x32_bf16(false, a.v, false, b.v,
                                                  (short)0, acc, false, false);
  }
  return acc;
}

// ----------------------------- pack / convert ------------------------------
__global__ void pack_x_kernel(const float* __restrict__ utt,
                              unsigned short* __restrict__ xbf) {
  size_t i = (size_t)blockIdx.x * blockDim.x + threadIdx.x;
  if (i >= (size_t)BB * SS * DD) return;
  int k = (int)(i % DD);
  size_t r = i / DD;
  int t = (int)(r % SS);
  int b = (int)(r / SS);
  xbf[((size_t)t * BB + b) * DD + k] = f2bf(utt[i]);   // (B,S,D) -> (S,B,D)
}

__global__ void cvt_bf16_kernel(const float* __restrict__ s,
                                unsigned short* __restrict__ d, int n) {
  int i = blockIdx.x * blockDim.x + threadIdx.x;
  if (i < n) d[i] = f2bf(s[i]);
}

__global__ void init_glob_kernel(const float* __restrict__ ig,
                                 float* __restrict__ gs) {
  int i = blockIdx.x * blockDim.x + threadIdx.x;
  if (i >= BB * SPN * HH) return;
  int j = i & (HH - 1);
  int sp = (i / HH) & (SPN - 1);
  gs[i] = ig[sp * HH + j];
}

// --------------------------- bulk WMMA GEMM --------------------------------
// C[M,N] = A[M,K] @ W[:, wcol:wcol+K].T + bias ; A,W bf16 row-major, C f32.
__global__ void __launch_bounds__(256)
gemm_bf16_kernel(const unsigned short* __restrict__ A, int lda,
                 const unsigned short* __restrict__ W, int ldw, int wcol,
                 const float* __restrict__ bias,
                 float* __restrict__ C, int ldc, int M, int N, int K) {
  int gwave = (int)((blockIdx.x * blockDim.x + threadIdx.x) >> 5);
  int lane  = threadIdx.x & 31;
  int ntn = N >> 4;
  int total = (M >> 4) * ntn;
  if (gwave >= total) return;
  int mt = gwave / ntn, nt = gwave % ntn;
  const unsigned short* arow = A + (size_t)(mt * 16 + (lane & 15)) * lda;
  const unsigned short* wrow = W + (size_t)(nt * 16 + (lane & 15)) * ldw + wcol;
  v8f acc = wmma_dot(arow, wrow, K, lane);
  int n = nt * 16 + (lane & 15);
  float bv = bias ? bias[n] : 0.f;
  int mb = mt * 16 + ((lane >= 16) ? 8 : 0);
#pragma unroll
  for (int v = 0; v < 8; ++v)
    C[(size_t)(mb + v) * ldc + n] = acc[v] + bv;
}

// ---------------------- global GRU recurrence (persistent) -----------------
// One workgroup, 512 threads = 16 waves. State of selected speaker gathered
// from global ws each step; gi/gh panels (16x1536 each) staged in LDS.
__global__ void __launch_bounds__(512)
global_gru_kernel(const float* __restrict__ gi_x,            // (S,B,3H) +bih
                  const unsigned short* __restrict__ wih_bf, // (3H,1536)
                  const unsigned short* __restrict__ whh_bf, // (3H,512)
                  const float* __restrict__ bhh,
                  const float* __restrict__ ln_g, const float* __restrict__ ln_b,
                  const int* __restrict__ spk_ids,           // (B,S)
                  float* __restrict__ gstate,                // (B,SP,H)
                  unsigned short* __restrict__ spk_out) {    // (S,B,H) bf16
  extern __shared__ char smem[];
  float* gih  = (float*)smem;                    // 16*1536
  float* ghh  = gih + BB * H3;                   // 16*1536
  float* hbuf = ghh + BB * H3;                   // 16*512
  unsigned short* hbf = (unsigned short*)(hbuf + BB * HH);  // 16*512 bf16
  const int tid = threadIdx.x, lane = tid & 31, wave = tid >> 5;

  for (int t = 0; t < SS; ++t) {
    // gather selected-speaker states
    for (int i = tid; i < BB * HH; i += 512) {
      int b = i >> 9, j = i & (HH - 1);
      int sp = spk_ids[b * SS + t];
      float hv = gstate[((size_t)b * SPN + sp) * HH + j];
      hbuf[i] = hv; hbf[i] = f2bf(hv);
    }
    __syncthreads();
    // gi_h = h @ wih[:,D:].T ; gh = h @ whh.T   (N total 3072 -> 192 tiles)
    for (int tt = wave; tt < 192; tt += 16) {
      int n0 = tt * 16;
      const unsigned short* wrow;
      float* dst; int dn0;
      if (n0 < H3) { wrow = wih_bf + (size_t)(n0 + (lane & 15)) * 1536 + DD;
                     dst = gih; dn0 = n0; }
      else         { wrow = whh_bf + (size_t)(n0 - H3 + (lane & 15)) * HH;
                     dst = ghh; dn0 = n0 - H3; }
      const unsigned short* arow = hbf + (lane & 15) * HH;
      v8f acc = wmma_dot(arow, wrow, HH, lane);
      int mb = (lane >= 16) ? 8 : 0, nn = dn0 + (lane & 15);
#pragma unroll
      for (int v = 0; v < 8; ++v) dst[(mb + v) * H3 + nn] = acc[v];
    }
    __syncthreads();
    // gates (r,z,n), blend, in-place update of hbuf
    for (int i = tid; i < BB * HH; i += 512) {
      int b = i >> 9, j = i & (HH - 1);
      const float* gx = gi_x + ((size_t)t * BB + b) * H3;
      float r = sigm(gih[b * H3 + j]        + gx[j]        + ghh[b * H3 + j]        + bhh[j]);
      float z = sigm(gih[b * H3 + HH + j]   + gx[HH + j]   + ghh[b * H3 + HH + j]   + bhh[HH + j]);
      float nn = tanhf(gih[b * H3 + 2*HH + j] + gx[2*HH + j]
                       + r * (ghh[b * H3 + 2*HH + j] + bhh[2*HH + j]));
      hbuf[i] = (1.f - z) * nn + z * hbuf[i];
    }
    __syncthreads();
    // LayerNorm per row: wave w <-> row b=w
    {
      int b = wave;
      float s = 0.f, s2 = 0.f;
      for (int j = lane; j < HH; j += 32) { float v = hbuf[b * HH + j]; s += v; s2 += v * v; }
      for (int off = 16; off > 0; off >>= 1) { s += __shfl_xor(s, off); s2 += __shfl_xor(s2, off); }
      float mean = s * (1.f / HH);
      float inv = rsqrtf(s2 * (1.f / HH) - mean * mean + 1e-5f);
      for (int j = lane; j < HH; j += 32) {
        float v = (hbuf[b * HH + j] - mean) * inv * ln_g[j] + ln_b[j];
        hbuf[b * HH + j] = v; hbf[b * HH + j] = f2bf(v);
      }
    }
    __syncthreads();
    // write back state + spk_state sequence (bf16, (S,B,H))
    for (int i = tid; i < BB * HH; i += 512) {
      int b = i >> 9, j = i & (HH - 1);
      int sp = spk_ids[b * SS + t];
      gstate[((size_t)b * SPN + sp) * HH + j] = hbuf[i];
      spk_out[((size_t)t * BB + b) * HH + j] = hbf[i];
    }
    __threadfence();
    __syncthreads();
  }
}

// ------------------- emotion GRU + output head (persistent) ----------------
__global__ void __launch_bounds__(512)
emotion_kernel(const float* __restrict__ gi_x,            // (S,B,3H) gi_es+bih
               const unsigned short* __restrict__ wih_bf, // eg_wih (3H,1024)
               const unsigned short* __restrict__ whh_bf, // (3H,512)
               const float* __restrict__ bhh,
               const float* __restrict__ ln_g, const float* __restrict__ ln_b,
               const float* __restrict__ init_emo,
               const unsigned short* __restrict__ ow1_bf,  // (256,512)
               const float* __restrict__ ob1,
               const float* __restrict__ olg, const float* __restrict__ olb,
               const float* __restrict__ ow2, const float* __restrict__ ob2,
               float* __restrict__ out) {                  // (B,S)
  extern __shared__ char smem[];
  float* gih  = (float*)smem;                               // 16*1536
  float* ghh  = gih + BB * H3;                              // 16*1536
  float* hbuf = ghh + BB * H3;                              // 16*512 emo f32
  unsigned short* hbf = (unsigned short*)(hbuf + BB * HH);  // 16*512 emo bf16
  unsigned short* ebf = hbf + BB * HH;                      // 16*512 e_out bf16
  float* h1 = (float*)(ebf + BB * HH);                      // 16*256
  const int tid = threadIdx.x, lane = tid & 31, wave = tid >> 5;

  for (int i = tid; i < BB * HH; i += 512) {
    float v = init_emo[i & (HH - 1)];
    hbuf[i] = v; hbf[i] = f2bf(v);
  }
  __syncthreads();

  for (int t = 0; t < SS; ++t) {
    // gi_h = emo @ eg_wih[:, :H].T ; gh = emo @ eg_whh.T
    for (int tt = wave; tt < 192; tt += 16) {
      int n0 = tt * 16;
      const unsigned short* wrow;
      float* dst; int dn0;
      if (n0 < H3) { wrow = wih_bf + (size_t)(n0 + (lane & 15)) * 1024;
                     dst = gih; dn0 = n0; }
      else         { wrow = whh_bf + (size_t)(n0 - H3 + (lane & 15)) * HH;
                     dst = ghh; dn0 = n0 - H3; }
      const unsigned short* arow = hbf + (lane & 15) * HH;
      v8f acc = wmma_dot(arow, wrow, HH, lane);
      int mb = (lane >= 16) ? 8 : 0, nn = dn0 + (lane & 15);
#pragma unroll
      for (int v = 0; v < 8; ++v) dst[(mb + v) * H3 + nn] = acc[v];
    }
    __syncthreads();
    // gates -> e_out (pre-LN kept in ebf for the head), state update in hbuf
    for (int i = tid; i < BB * HH; i += 512) {
      int b = i >> 9, j = i & (HH - 1);
      const float* gx = gi_x + ((size_t)t * BB + b) * H3;
      float r = sigm(gih[b * H3 + j]      + gx[j]      + ghh[b * H3 + j]      + bhh[j]);
      float z = sigm(gih[b * H3 + HH + j] + gx[HH + j] + ghh[b * H3 + HH + j] + bhh[HH + j]);
      float nn = tanhf(gih[b * H3 + 2*HH + j] + gx[2*HH + j]
                       + r * (ghh[b * H3 + 2*HH + j] + bhh[2*HH + j]));
      float e = (1.f - z) * nn + z * hbuf[i];
      hbuf[i] = e; ebf[i] = f2bf(e);
    }
    __syncthreads();
    // LN of emotion state (carry)
    {
      int b = wave;
      float s = 0.f, s2 = 0.f;
      for (int j = lane; j < HH; j += 32) { float v = hbuf[b * HH + j]; s += v; s2 += v * v; }
      for (int off = 16; off > 0; off >>= 1) { s += __shfl_xor(s, off); s2 += __shfl_xor(s2, off); }
      float mean = s * (1.f / HH);
      float inv = rsqrtf(s2 * (1.f / HH) - mean * mean + 1e-5f);
      for (int j = lane; j < HH; j += 32) {
        float v = (hbuf[b * HH + j] - mean) * inv * ln_g[j] + ln_b[j];
        hbuf[b * HH + j] = v; hbf[b * HH + j] = f2bf(v);
      }
    }
    __syncthreads();
    // head GEMM: h1 = relu(e_out @ out_w1.T + b1)  (N=256 -> 16 tiles)
    {
      int n0 = wave * 16;
      const unsigned short* arow = ebf + (lane & 15) * HH;
      const unsigned short* wrow = ow1_bf + (size_t)(n0 + (lane & 15)) * HH;
      v8f acc = wmma_dot(arow, wrow, HH, lane);
      int mb = (lane >= 16) ? 8 : 0, nn = n0 + (lane & 15);
#pragma unroll
      for (int v = 0; v < 8; ++v)
        h1[(mb + v) * 256 + nn] = fmaxf(acc[v] + ob1[nn], 0.f);
    }
    __syncthreads();
    // LN over 256 + score = sigmoid(h1n @ out_w2.T + b2)
    {
      int b = wave;
      float s = 0.f, s2 = 0.f;
      for (int j = lane; j < 256; j += 32) { float v = h1[b * 256 + j]; s += v; s2 += v * v; }
      for (int off = 16; off > 0; off >>= 1) { s += __shfl_xor(s, off); s2 += __shfl_xor(s2, off); }
      float mean = s * (1.f / 256.f);
      float inv = rsqrtf(s2 * (1.f / 256.f) - mean * mean + 1e-5f);
      float dot = 0.f;
      for (int j = lane; j < 256; j += 32) {
        float v = (h1[b * 256 + j] - mean) * inv * olg[j] + olb[j];
        dot += v * ow2[j];
      }
      for (int off = 16; off > 0; off >>= 1) dot += __shfl_xor(dot, off);
      if (lane == 0) out[(size_t)b * SS + t] = sigm(dot + ob2[0]);
    }
    __syncthreads();
  }
}

// ------------------------------- launcher ----------------------------------
extern "C" void kernel_launch(void* const* d_in, const int* in_sizes, int n_in,
                              void* d_out, int out_size, void* d_ws, size_t ws_size,
                              hipStream_t stream) {
  (void)in_sizes; (void)n_in; (void)out_size; (void)ws_size;
  const float* utt       = (const float*)d_in[0];
  const int*   spk       = (const int*)d_in[1];
  const float* init_glob = (const float*)d_in[3];
  const float* init_emo  = (const float*)d_in[4];
  const float* gg_wih    = (const float*)d_in[9];
  const float* gg_whh    = (const float*)d_in[10];
  const float* gg_bih    = (const float*)d_in[11];
  const float* gg_bhh    = (const float*)d_in[12];
  const float* eg_wih    = (const float*)d_in[13];
  const float* eg_whh    = (const float*)d_in[14];
  const float* eg_bih    = (const float*)d_in[15];
  const float* eg_bhh    = (const float*)d_in[16];
  const float* ln_g      = (const float*)d_in[20];
  const float* ln_b      = (const float*)d_in[21];
  const float* out_w1    = (const float*)d_in[22];
  const float* out_b1    = (const float*)d_in[23];
  const float* oln_g     = (const float*)d_in[24];
  const float* oln_b     = (const float*)d_in[25];
  const float* out_w2    = (const float*)d_in[26];
  const float* out_b2    = (const float*)d_in[27];

  // workspace layout (bytes)
  char* ws = (char*)d_ws;
  const size_t OFF_XBF   = 0;                                    // 8192*1024*2
  const size_t OFF_GGWIH = OFF_XBF   + (size_t)SS*BB*DD*2;       // 1536*1536*2
  const size_t OFF_GGWHH = OFF_GGWIH + (size_t)H3*1536*2;        // 1536*512*2
  const size_t OFF_EGWIH = OFF_GGWHH + (size_t)H3*HH*2;          // 1536*1024*2
  const size_t OFF_EGWHH = OFF_EGWIH + (size_t)H3*1024*2;        // 1536*512*2
  const size_t OFF_OW1   = OFF_EGWHH + (size_t)H3*HH*2;          // 256*512*2
  const size_t OFF_SPKBF = OFF_OW1   + (size_t)256*HH*2;         // 8192*512*2
  const size_t OFF_GST   = OFF_SPKBF + (size_t)SS*BB*HH*2;       // 16*8*512*4
  const size_t OFF_GI    = OFF_GST   + (size_t)BB*SPN*HH*4;      // 8192*1536*4 (reused)

  unsigned short* xbf    = (unsigned short*)(ws + OFF_XBF);
  unsigned short* ggwihb = (unsigned short*)(ws + OFF_GGWIH);
  unsigned short* ggwhhb = (unsigned short*)(ws + OFF_GGWHH);
  unsigned short* egwihb = (unsigned short*)(ws + OFF_EGWIH);
  unsigned short* egwhhb = (unsigned short*)(ws + OFF_EGWHH);
  unsigned short* ow1b   = (unsigned short*)(ws + OFF_OW1);
  unsigned short* spkbf  = (unsigned short*)(ws + OFF_SPKBF);
  float* gstate          = (float*)(ws + OFF_GST);
  float* gi_buf          = (float*)(ws + OFF_GI);   // gi_gx, then gi_es

  // 1) pack & convert
  {
    size_t n = (size_t)BB * SS * DD;
    pack_x_kernel<<<(unsigned)((n + 255) / 256), 256, 0, stream>>>(utt, xbf);
  }
  cvt_bf16_kernel<<<(H3*1536 + 255)/256, 256, 0, stream>>>(gg_wih, ggwihb, H3*1536);
  cvt_bf16_kernel<<<(H3*HH   + 255)/256, 256, 0, stream>>>(gg_whh, ggwhhb, H3*HH);
  cvt_bf16_kernel<<<(H3*1024 + 255)/256, 256, 0, stream>>>(eg_wih, egwihb, H3*1024);
  cvt_bf16_kernel<<<(H3*HH   + 255)/256, 256, 0, stream>>>(eg_whh, egwhhb, H3*HH);
  cvt_bf16_kernel<<<(256*HH  + 255)/256, 256, 0, stream>>>(out_w1, ow1b, 256*HH);

  // 2) gi_gx = Xbf @ gg_wih[:, :D].T + gg_bih   (M=8192,N=1536,K=1024)
  {
    int tiles = (SS*BB/16) * (H3/16);
    gemm_bf16_kernel<<<(tiles + 7)/8, 256, 0, stream>>>(
        xbf, DD, ggwihb, 1536, 0, gg_bih, gi_buf, H3, SS*BB, H3, DD);
  }

  // 3) global GRU recurrence
  init_glob_kernel<<<(BB*SPN*HH + 255)/256, 256, 0, stream>>>(init_glob, gstate);
  {
    size_t smem3 = (size_t)(2*BB*H3 + BB*HH)*4 + (size_t)BB*HH*2;   // 240 KB
    hipFuncSetAttribute((const void*)global_gru_kernel,
                        hipFuncAttributeMaxDynamicSharedMemorySize, (int)smem3);
    global_gru_kernel<<<1, 512, smem3, stream>>>(
        gi_buf, ggwihb, ggwhhb, gg_bhh, ln_g, ln_b, spk, gstate, spkbf);
  }

  // 4) gi_es = spk_state_bf @ eg_wih[:, H:].T + eg_bih (overwrites gi_buf)
  {
    int tiles = (SS*BB/16) * (H3/16);
    gemm_bf16_kernel<<<(tiles + 7)/8, 256, 0, stream>>>(
        spkbf, HH, egwihb, 1024, HH, eg_bih, gi_buf, H3, SS*BB, H3, HH);
  }

  // 5) emotion GRU + head -> scores (B,S,1)
  {
    size_t smem4 = (size_t)(2*BB*H3 + BB*HH)*4 + (size_t)2*BB*HH*2
                 + (size_t)BB*256*4;                                // 272 KB
    hipFuncSetAttribute((const void*)emotion_kernel,
                        hipFuncAttributeMaxDynamicSharedMemorySize, (int)smem4);
    emotion_kernel<<<1, 512, smem4, stream>>>(
        gi_buf, egwihb, egwhhb, eg_bhh, ln_g, ln_b, init_emo,
        ow1b, out_b1, oln_g, oln_b, out_w2, out_b2, (float*)d_out);
  }
}
